// Net_12816182411733
// MI455X (gfx1250) — compile-verified, matrix operand order checked by reading, not run on hardware
//
#include <hip/hip_runtime.h>
#include <hip/hip_bf16.h>
#include <stdint.h>

typedef _Float16 h16;
typedef _Float16 v8h  __attribute__((ext_vector_type(8)));
typedef _Float16 v16h __attribute__((ext_vector_type(16)));
typedef float    v8f  __attribute__((ext_vector_type(8)));
typedef float    v4f  __attribute__((ext_vector_type(4)));

#define IN_K    784
#define HID     512
#define BM      128         // batch rows per workgroup (8 m-tiles)
#define NTHREADS 512        // 16 waves
#define SPITCH  520         // LDS activation row pitch (halves); breaks bank-conflict stride

constexpr size_t SZ_L1  = 25ull * 32 * 512;   // K=800(pad of 784), N=512
constexpr size_t SZ_MID = 16ull * 32 * 512;   // K=512, N=512
constexpr size_t OFF1 = 0;
constexpr size_t OFF2 = OFF1 + SZ_L1;
constexpr size_t OFF3 = OFF2 + SZ_MID;
constexpr size_t OFF4 = OFF3 + SZ_MID;
constexpr size_t OFF5 = OFF4 + SZ_MID;
constexpr size_t OFF6 = OFF5 + SZ_MID;
constexpr size_t OFF7 = OFF6 + SZ_MID;
constexpr size_t OFF8 = OFF7 + SZ_MID;        // K=512, N=16(pad of 10)

// ---------------------------------------------------------------------------
// Repack f32 row-major (K x N) weights into fp16 WMMA B-fragment order:
// tile (kt,nt) = 512 halves; half index = lane*16 + j with
//   k = kt*32 + (j<8 ? j : j+8) + (lane>=16 ? 8 : 0)
//   n = nt*16 + (lane & 15)
// so the GEMM reads one contiguous 32B chunk per lane per fragment.
// ---------------------------------------------------------------------------
__global__ void pack_weights(const float* __restrict__ W, h16* __restrict__ dst,
                             int Kreal, int Ktiles, int Nreal, int Ntiles) {
  int idx   = blockIdx.x * 256 + threadIdx.x;
  int total = Ktiles * Ntiles * 512;
  if (idx >= total) return;
  int t    = idx & 511;
  int tile = idx >> 9;
  int nt   = tile % Ntiles;
  int kt   = tile / Ntiles;
  int lane = t >> 4;
  int j    = t & 15;
  int k = kt * 32 + (j < 8 ? j : j + 8) + ((lane & 16) ? 8 : 0);
  int n = nt * 16 + (lane & 15);
  float v = (k < Kreal && n < Nreal) ? W[(size_t)k * Nreal + n] : 0.0f;
  dst[idx] = (h16)v;
}

// ---------------------------------------------------------------------------
static __device__ __forceinline__ float hash01(uint32_t x) {
  x ^= x >> 17; x *= 0xed5ad4bbu;
  x ^= x >> 11; x *= 0xac4c1b51u;
  x ^= x >> 15; x *= 0x31848babu;
  x ^= x >> 14;
  return (float)(x >> 8) * (1.0f / 16777216.0f);
}

static __device__ __forceinline__ v8f zero8() {
  v8f z;
#pragma unroll
  for (int i = 0; i < 8; ++i) z[i] = 0.0f;
  return z;
}

static __device__ __forceinline__ v8h loadcvt8(const float* __restrict__ p, int k) {
  v8h r;
  if (k + 8 <= IN_K) {
    v4f f0 = *(const v4f*)(p + k);
    v4f f1 = *(const v4f*)(p + k + 4);
#pragma unroll
    for (int i = 0; i < 4; ++i) { r[i] = (h16)f0[i]; r[4 + i] = (h16)f1[i]; }
  } else {
#pragma unroll
    for (int i = 0; i < 8; ++i) r[i] = (h16)0.0f;
  }
  return r;
}

// Assemble a 16-half A fragment from LDS (two 16B ds_load_b128).
static __device__ __forceinline__ v16h ldsA(const h16* sIn, int row, int kb) {
  const h16* p = sIn + row * SPITCH + kb;
  v8h lo = *(const v8h*)p;
  v8h hh = *(const v8h*)(p + 16);
  v16h a;
#pragma unroll
  for (int i = 0; i < 8; ++i) { a[i] = lo[i]; a[8 + i] = hh[i]; }
  return a;
}

// ---------------------------------------------------------------------------
// Layer 1: A-fragments straight from global x (f32 -> f16), K padded to 800.
// m-tiles processed in pairs: 4 back-to-back WMMAs per A-register overwrite,
// halving the WMMA->VALU WAR hazard NOP overhead.
// ---------------------------------------------------------------------------
static __device__ __forceinline__ void layer1(
    const float* __restrict__ x, h16* sOut, const h16* __restrict__ Wp,
    const float* __restrict__ bias, int rowBase) {
  const int tid   = threadIdx.x;
  const int wave  = tid >> 5;
  const int lane  = tid & 31;
  const int laneN = lane & 15;
  const int hi    = lane >> 4;
  const int ntB   = wave * 2;

  v8f acc[8][2];
#pragma unroll
  for (int mt = 0; mt < 8; ++mt) { acc[mt][0] = zero8(); acc[mt][1] = zero8(); }

  for (int kt = 0; kt < 25; ++kt) {
    const int kb = kt * 32 + hi * 8;
    v16h b0 = *(const v16h*)(Wp + ((size_t)(kt * 32 + ntB + 0) * 512) + lane * 16);
    v16h b1 = *(const v16h*)(Wp + ((size_t)(kt * 32 + ntB + 1) * 512) + lane * 16);
#pragma unroll
    for (int mt = 0; mt < 8; mt += 2) {
      const float* xr0 = x + (size_t)(rowBase + (mt + 0) * 16 + laneN) * IN_K;
      const float* xr1 = x + (size_t)(rowBase + (mt + 1) * 16 + laneN) * IN_K;
      v8h lo0 = loadcvt8(xr0, kb), hh0 = loadcvt8(xr0, kb + 16);
      v8h lo1 = loadcvt8(xr1, kb), hh1 = loadcvt8(xr1, kb + 16);
      v16h a0, a1;
#pragma unroll
      for (int i = 0; i < 8; ++i) {
        a0[i] = lo0[i]; a0[8 + i] = hh0[i];
        a1[i] = lo1[i]; a1[8 + i] = hh1[i];
      }
      acc[mt + 0][0] = __builtin_amdgcn_wmma_f32_16x16x32_f16(
          false, a0, false, b0, (short)0, acc[mt + 0][0], false, false);
      acc[mt + 0][1] = __builtin_amdgcn_wmma_f32_16x16x32_f16(
          false, a0, false, b1, (short)0, acc[mt + 0][1], false, false);
      acc[mt + 1][0] = __builtin_amdgcn_wmma_f32_16x16x32_f16(
          false, a1, false, b0, (short)0, acc[mt + 1][0], false, false);
      acc[mt + 1][1] = __builtin_amdgcn_wmma_f32_16x16x32_f16(
          false, a1, false, b1, (short)0, acc[mt + 1][1], false, false);
    }
  }

#pragma unroll
  for (int nt = 0; nt < 2; ++nt) {
    const int n = (ntB + nt) * 16 + laneN;
    const float bv = bias[n];
#pragma unroll
    for (int mt = 0; mt < 8; ++mt)
#pragma unroll
      for (int i = 0; i < 8; ++i) {
        const int r = mt * 16 + hi * 8 + i;
        float v = fmaxf(acc[mt][nt][i] + bv, 0.0f);
        sOut[r * SPITCH + n] = (h16)v;
      }
  }
}

// ---------------------------------------------------------------------------
// Middle layers (K=512, N=512), LDS in / LDS out, optional fused dropout.
// ---------------------------------------------------------------------------
template <bool DROP>
static __device__ __forceinline__ void layer_mid(
    const h16* sIn, h16* sOut, const h16* __restrict__ Wp,
    const float* __restrict__ bias, float keep, uint32_t seed, int rowBase) {
  const int tid   = threadIdx.x;
  const int wave  = tid >> 5;
  const int lane  = tid & 31;
  const int laneN = lane & 15;
  const int hi    = lane >> 4;
  const int ntB   = wave * 2;

  v8f acc[8][2];
#pragma unroll
  for (int mt = 0; mt < 8; ++mt) { acc[mt][0] = zero8(); acc[mt][1] = zero8(); }

  for (int kt = 0; kt < 16; ++kt) {
    const int kb = kt * 32 + hi * 8;
    if (kt + 1 < 16)  // prefetch next weight k-slab (global_prefetch_b8)
      __builtin_prefetch(Wp + ((size_t)((kt + 1) * 32 + ntB) * 512) + lane * 16, 0, 0);
    v16h b0 = *(const v16h*)(Wp + ((size_t)(kt * 32 + ntB + 0) * 512) + lane * 16);
    v16h b1 = *(const v16h*)(Wp + ((size_t)(kt * 32 + ntB + 1) * 512) + lane * 16);
#pragma unroll
    for (int mt = 0; mt < 8; mt += 2) {
      v16h a0 = ldsA(sIn, (mt + 0) * 16 + laneN, kb);
      v16h a1 = ldsA(sIn, (mt + 1) * 16 + laneN, kb);
      acc[mt + 0][0] = __builtin_amdgcn_wmma_f32_16x16x32_f16(
          false, a0, false, b0, (short)0, acc[mt + 0][0], false, false);
      acc[mt + 0][1] = __builtin_amdgcn_wmma_f32_16x16x32_f16(
          false, a0, false, b1, (short)0, acc[mt + 0][1], false, false);
      acc[mt + 1][0] = __builtin_amdgcn_wmma_f32_16x16x32_f16(
          false, a1, false, b0, (short)0, acc[mt + 1][0], false, false);
      acc[mt + 1][1] = __builtin_amdgcn_wmma_f32_16x16x32_f16(
          false, a1, false, b1, (short)0, acc[mt + 1][1], false, false);
    }
  }

  const float invkeep = DROP ? (1.0f / keep) : 1.0f;
#pragma unroll
  for (int nt = 0; nt < 2; ++nt) {
    const int n = (ntB + nt) * 16 + laneN;
    const float bv = bias[n];
#pragma unroll
    for (int mt = 0; mt < 8; ++mt)
#pragma unroll
      for (int i = 0; i < 8; ++i) {
        const int r = mt * 16 + hi * 8 + i;
        float v = fmaxf(acc[mt][nt][i] + bv, 0.0f);
        if (DROP) {
          uint32_t id = (uint32_t)(rowBase + r) * 512u + (uint32_t)n;
          float u = hash01(id * 2654435761u + seed);
          v = (u < keep) ? v * invkeep : 0.0f;
        }
        sOut[r * SPITCH + n] = (h16)v;
      }
  }
}

// ---------------------------------------------------------------------------
// Layer 8: 512 -> 10 (padded 16). Waves 0-1 handle the 8 m-tiles.
// ---------------------------------------------------------------------------
static __device__ __forceinline__ void layer8(
    const h16* sIn, float* sLog, const h16* __restrict__ Wp,
    const float* __restrict__ b8) {
  const int tid  = threadIdx.x;
  const int wave = tid >> 5;
  if (wave >= 2) return;
  const int lane  = tid & 31;
  const int laneN = lane & 15;
  const int hi    = lane >> 4;
  const int mtB   = wave * 4;

  v8f acc[4];
#pragma unroll
  for (int mt = 0; mt < 4; ++mt) acc[mt] = zero8();

  for (int kt = 0; kt < 16; ++kt) {
    const int kb = kt * 32 + hi * 8;
    v16h b = *(const v16h*)(Wp + (size_t)kt * 512 + lane * 16);
#pragma unroll
    for (int mt = 0; mt < 4; mt += 2) {
      v16h a0 = ldsA(sIn, (mtB + mt + 0) * 16 + laneN, kb);
      v16h a1 = ldsA(sIn, (mtB + mt + 1) * 16 + laneN, kb);
      acc[mt + 0] = __builtin_amdgcn_wmma_f32_16x16x32_f16(
          false, a0, false, b, (short)0, acc[mt + 0], false, false);
      acc[mt + 1] = __builtin_amdgcn_wmma_f32_16x16x32_f16(
          false, a1, false, b, (short)0, acc[mt + 1], false, false);
    }
  }
  const float bv = (laneN < 10) ? b8[laneN] : 0.0f;
#pragma unroll
  for (int mt = 0; mt < 4; ++mt)
#pragma unroll
    for (int i = 0; i < 8; ++i)
      sLog[((mtB + mt) * 16 + hi * 8 + i) * 16 + laneN] = acc[mt][i] + bv;
}

// ---------------------------------------------------------------------------
__global__ __launch_bounds__(NTHREADS)
void mlp_fused(const float* __restrict__ x, const h16* __restrict__ wp,
               const float* __restrict__ b1, const float* __restrict__ b2,
               const float* __restrict__ b3, const float* __restrict__ b4,
               const float* __restrict__ b5, const float* __restrict__ b6,
               const float* __restrict__ b7, const float* __restrict__ b8,
               float* __restrict__ out) {
  __shared__ h16 sA[BM * SPITCH];   // 133,120 B
  __shared__ h16 sB[BM * SPITCH];   // 133,120 B  (total ~260 KB of 320 KB WGP LDS)
  const int rowBase = blockIdx.x * BM;

  layer1(x, sA, wp + OFF1, b1, rowBase);                                  __syncthreads();
  layer_mid<true >(sA, sB, wp + OFF2, b2, 0.8f, 0xA5F152E1u, rowBase);    __syncthreads();
  layer_mid<false>(sB, sA, wp + OFF3, b3, 1.0f, 0u,          rowBase);    __syncthreads();
  layer_mid<true >(sA, sB, wp + OFF4, b4, 0.7f, 0x51E7B007u, rowBase);    __syncthreads();
  layer_mid<false>(sB, sA, wp + OFF5, b5, 1.0f, 0u,          rowBase);    __syncthreads();
  layer_mid<true >(sA, sB, wp + OFF6, b6, 0.5f, 0xC0FFEE11u, rowBase);    __syncthreads();
  layer_mid<false>(sB, sA, wp + OFF7, b7, 1.0f, 0u,          rowBase);    __syncthreads();

  float* sLog = (float*)sB;   // 128 x 16 f32 logits, reuse ping buffer
  layer8(sA, sLog, wp + OFF8, b8);
  __syncthreads();

  if (threadIdx.x < BM) {
    const float* lr = sLog + threadIdx.x * 16;
    float mx = lr[0];
#pragma unroll
    for (int c = 1; c < 10; ++c) mx = fmaxf(mx, lr[c]);
    float e[10], s = 0.0f;
#pragma unroll
    for (int c = 0; c < 10; ++c) { e[c] = __expf(lr[c] - mx); s += e[c]; }
    const float inv = 1.0f / s;
    float* o = out + (size_t)(rowBase + threadIdx.x) * 10;
#pragma unroll
    for (int c = 0; c < 10; ++c) o[c] = e[c] * inv;
  }
}

// ---------------------------------------------------------------------------
extern "C" void kernel_launch(void* const* d_in, const int* in_sizes, int n_in,
                              void* d_out, int out_size, void* d_ws, size_t ws_size,
                              hipStream_t stream) {
  (void)in_sizes; (void)n_in; (void)out_size; (void)ws_size;
  const float* x = (const float*)d_in[0];
  const float* W[8]; const float* b[8];
  for (int i = 0; i < 8; ++i) {
    W[i] = (const float*)d_in[1 + 2 * i];
    b[i] = (const float*)d_in[2 + 2 * i];
  }
  h16* wp = (h16*)d_ws;   // ~3.98 MB of packed fp16 weights

  auto pack = [&](const float* Wl, h16* dst, int Kreal, int Ktiles, int Nreal, int Ntiles) {
    int total = Ktiles * Ntiles * 512;
    pack_weights<<<(total + 255) / 256, 256, 0, stream>>>(Wl, dst, Kreal, Ktiles, Nreal, Ntiles);
  };
  pack(W[0], wp + OFF1, 784, 25, 512, 32);
  pack(W[1], wp + OFF2, 512, 16, 512, 32);
  pack(W[2], wp + OFF3, 512, 16, 512, 32);
  pack(W[3], wp + OFF4, 512, 16, 512, 32);
  pack(W[4], wp + OFF5, 512, 16, 512, 32);
  pack(W[5], wp + OFF6, 512, 16, 512, 32);
  pack(W[6], wp + OFF7, 512, 16, 512, 32);
  pack(W[7], wp + OFF8, 512, 16, 10, 1);

  mlp_fused<<<65536 / BM, NTHREADS, 0, stream>>>(
      x, wp, b[0], b[1], b[2], b[3], b[4], b[5], b[6], b[7], (float*)d_out);
}
